// W8A8LinearAIU_26096221290711
// MI455X (gfx1250) — 8-bit matrix operand order repaired and checked by reading, compiled, not run on hardware
//
#include <hip/hip_runtime.h>
#include <stdint.h>

typedef __attribute__((ext_vector_type(2))) int   v2i;
typedef __attribute__((ext_vector_type(4))) int   v4i;
typedef __attribute__((ext_vector_type(8))) int   v8i;
typedef __attribute__((ext_vector_type(4))) float v4f;

#define M_TOK (8 * 4096)
#define K_DIM 1024
#define N_DIM 1024

__global__ __launch_bounds__(256) void w8a8_quant_kernel(
    const float* __restrict__ x,
    const float* __restrict__ smooth,
    int*        __restrict__ xq_packed,
    float*      __restrict__ a_scale_out)
{
    __shared__ float red[8];
    const int row = blockIdx.x;
    const int tid = threadIdx.x;

    const v4f xv = ((const v4f*)(x + (size_t)row * K_DIM))[tid];
    const v4f sv = ((const v4f*)smooth)[tid];

    v4f xs;
    xs[0] = xv[0] / sv[0];
    xs[1] = xv[1] / sv[1];
    xs[2] = xv[2] / sv[2];
    xs[3] = xv[3] / sv[3];

    float m = fmaxf(fmaxf(fabsf(xs[0]), fabsf(xs[1])),
                    fmaxf(fabsf(xs[2]), fabsf(xs[3])));
    #pragma unroll
    for (int off = 16; off > 0; off >>= 1)
        m = fmaxf(m, __shfl_xor(m, off, 32));

    if ((tid & 31) == 0) red[tid >> 5] = m;
    __syncthreads();
    if (tid == 0) {
        float mm = red[0];
        #pragma unroll
        for (int i = 1; i < 8; ++i) mm = fmaxf(mm, red[i]);
        red[0] = fmaxf(mm * (1.0f / 127.0f), 1e-8f);
    }
    __syncthreads();

    const float asc = red[0];
    const float inv = 1.0f / asc;

    int q[4];
    #pragma unroll
    for (int i = 0; i < 4; ++i) {
        float v = xs[i] * inv;
        v = fminf(fmaxf(v, -127.0f), 127.0f);
        q[i] = (int)rintf(v);
    }
    const int packed = (q[0] & 255) | ((q[1] & 255) << 8) |
                       ((q[2] & 255) << 16) | ((q[3] & 255) << 24);
    xq_packed[(size_t)row * (K_DIM / 4) + tid] = packed;
    if (tid == 0) a_scale_out[row] = asc;
}

__device__ __forceinline__ v8i load_a_frag(const int8_t* p) {
    v2i a0 = *(const v2i*)(p +  0);
    v2i a1 = *(const v2i*)(p + 16);
    v2i a2 = *(const v2i*)(p + 32);
    v2i a3 = *(const v2i*)(p + 48);
    v8i A;
    A[0] = a0[0]; A[1] = a0[1]; A[2] = a1[0]; A[3] = a1[1];
    A[4] = a2[0]; A[5] = a2[1]; A[6] = a3[0]; A[7] = a3[1];
    return A;
}

__device__ __forceinline__ v8i load_b_frag(const int8_t* p) {
    v2i b0 = *(const v2i*)(p +  0);
    v2i b1 = *(const v2i*)(p + 16);
    v2i b2 = *(const v2i*)(p + 32);
    v2i b3 = *(const v2i*)(p + 48);
    v8i B;
    B[0] = b0[0]; B[1] = b0[1]; B[2] = b1[0]; B[3] = b1[1];
    B[4] = b2[0]; B[5] = b2[1]; B[6] = b3[0]; B[7] = b3[1];
    return B;
}

__global__ __launch_bounds__(256) void w8a8_gemm_kernel(
    const int8_t* __restrict__ xq,
    const int8_t* __restrict__ w,
    const float*  __restrict__ a_scale,
    const float*  __restrict__ w_clip,
    const float*  __restrict__ bias,
    float*        __restrict__ out)
{
    const int tid    = threadIdx.x;
    const int lane   = tid & 31;
    const int wv     = tid >> 5;
    const int wm     = wv >> 2;
    const int wn     = wv & 3;
    const int m_wave = blockIdx.x * 64 + wm * 32;
    const int n_wave = blockIdx.y * 128 + wn * 32;
    const int col    = lane & 15;
    const int grp    = lane >> 4;

    const int8_t* arow0 = xq + (size_t)(m_wave + col) * K_DIM + grp * 8;
    const int8_t* arow1 = arow0 + (size_t)16 * K_DIM;
    const int8_t* brow0 = w  + (size_t)(n_wave + col) * K_DIM + grp * 8;
    const int8_t* brow1 = brow0 + (size_t)16 * K_DIM;

    v8i acc00 = {}, acc01 = {}, acc10 = {}, acc11 = {};

    for (int k = 0; k < K_DIM; k += 64) {
        v8i A0 = load_a_frag(arow0 + k);
        v8i A1 = load_a_frag(arow1 + k);
        v8i B0 = load_b_frag(brow0 + k);
        v8i B1 = load_b_frag(brow1 + k);

        acc00 = __builtin_amdgcn_wmma_i32_16x16x64_iu8(true, A0, true, B0,
                                                       acc00, false, false);
        acc01 = __builtin_amdgcn_wmma_i32_16x16x64_iu8(true, A0, true, B1,
                                                       acc01, false, false);
        acc10 = __builtin_amdgcn_wmma_i32_16x16x64_iu8(true, A1, true, B0,
                                                       acc10, false, false);
        acc11 = __builtin_amdgcn_wmma_i32_16x16x64_iu8(true, A1, true, B1,
                                                       acc11, false, false);
    }

    const int   n0   = n_wave + col;
    const int   n1   = n0 + 16;
    const float wsc0 = w_clip[n0] * (1.0f / 127.0f);
    const float wsc1 = w_clip[n1] * (1.0f / 127.0f);
    const float bi0  = bias[n0];
    const float bi1  = bias[n1];

    #pragma unroll
    for (int j = 0; j < 8; ++j) {
        const int   m0  = m_wave + grp * 8 + j;
        const int   m1  = m0 + 16;
        const float as0 = a_scale[m0];
        const float as1 = a_scale[m1];
        out[(size_t)m0 * N_DIM + n0] = (float)acc00[j] * as0 * wsc0 + bi0;
        out[(size_t)m0 * N_DIM + n1] = (float)acc01[j] * as0 * wsc1 + bi1;
        out[(size_t)m1 * N_DIM + n0] = (float)acc10[j] * as1 * wsc0 + bi0;
        out[(size_t)m1 * N_DIM + n1] = (float)acc11[j] * as1 * wsc1 + bi1;
    }
}

extern "C" void kernel_launch(void* const* d_in, const int* in_sizes, int n_in,
                              void* d_out, int out_size, void* d_ws, size_t ws_size,
                              hipStream_t stream)
{
    const float*  x      = (const float*)d_in[0];
    const int8_t* weight = (const int8_t*)d_in[1];
    const float*  bias   = (const float*)d_in[2];
    const float*  w_clip = (const float*)d_in[3];
    const float*  smooth = (const float*)d_in[4];
    float*        out    = (float*)d_out;

    int8_t* xq      = (int8_t*)d_ws;
    float*  a_scale = (float*)((char*)d_ws + (size_t)M_TOK * K_DIM);

    w8a8_quant_kernel<<<M_TOK, 256, 0, stream>>>(x, smooth, (int*)xq, a_scale);

    dim3 grid(M_TOK / 64, N_DIM / 128);
    w8a8_gemm_kernel<<<grid, 256, 0, stream>>>(xq, weight, a_scale, w_clip,
                                               bias, out);
}
